// GmmFull_137438953787
// MI455X (gfx1250) — matrix operand order repaired and checked by reading, not compiled, not run on hardware
//
#include <hip/hip_runtime.h>

// ---------------------------------------------------------------------------
// GMM full-covariance NLL on MI455X (gfx1250, wave32).
//
// Strategy:
//   setup kernel : A_k = L_k^{-1} (forward substitution in LDS),
//                  v_k = A_k mu_k, cadd_k = logsoftmax_k - logdet_k - D/2*log2pi,
//                  A_k pre-swizzled into WMMA bf16 A-fragment layout (hi+lo split).
//   main kernel  : per 16-row tile, Z = A_k X via v_wmma_f32_16x16x32_bf16
//                  (3-term hi/lo split for ~fp32 accuracy),
//                  maha via ds_swizzle lane-pair fold, streaming logsumexp over k.
//   final kernel : deterministic sequential sum of block partials -> -mean(lp).
// ---------------------------------------------------------------------------

#define D_DIM   64
#define K_COMP  32
#define LOG2PI  1.8378770664093453f

typedef __attribute__((ext_vector_type(16))) __bf16 v16bf;
typedef __attribute__((ext_vector_type(8)))  float  v8f;

__device__ __forceinline__ unsigned short f32_to_bf16(float f) {
  unsigned u = __builtin_bit_cast(unsigned, f);
  unsigned r = u + 0x7FFFu + ((u >> 16) & 1u);   // round-to-nearest-even
  return (unsigned short)(r >> 16);
}
__device__ __forceinline__ float bf16_to_f32(unsigned short b) {
  unsigned u = ((unsigned)b) << 16;
  return __builtin_bit_cast(float, u);
}
// ds_swizzle needs an immediate offset -> template parameter.
template <int OFF>
__device__ __forceinline__ float swz(float v) {
  return __builtin_bit_cast(float,
      __builtin_amdgcn_ds_swizzle(__builtin_bit_cast(int, v), OFF));
}

// ---------------------------------------------------------------------------
// Setup: one block per component k, 64 threads (thread j owns column j of A_k).
// aws layout (bf16): [k][part(hi/lo)][m(z-block 0..3)][ks(0..1)][lane 0..31][e 0..15]
//   -> flat element index idx = part*4096 + m*1024 + ks*512 + lane*16 + e,
//      element = Ainv[16m + lane%16][ks*32 + (lane/16)*16 + e]
//   i.e. contiguous k = 16*(lane/16) + e order, the same order as the B fragments.
// ---------------------------------------------------------------------------
__global__ __launch_bounds__(64) void gmm_setup_kernel(
    const float* __restrict__ mus, const float* __restrict__ Lmat,
    const float* __restrict__ logits,
    unsigned short* __restrict__ aws, float* __restrict__ vws,
    float* __restrict__ cws) {
  __shared__ float ldsL[D_DIM * 65];   // L row-major, padded (row*65 + col)
  __shared__ float ldsA[D_DIM * 65];   // Ainv column-major, padded (col*65 + row)
  const int k = blockIdx.x;
  const int j = threadIdx.x;

  // load L_k row j
  for (int m = 0; m < D_DIM; ++m)
    ldsL[j * 65 + m] = Lmat[(size_t)k * D_DIM * D_DIM + j * D_DIM + m];
  __syncthreads();

  // forward substitution: column j of Ainv solving L * a = e_j
  for (int i = 0; i < j; ++i) ldsA[j * 65 + i] = 0.0f;
  ldsA[j * 65 + j] = 1.0f / ldsL[j * 65 + j];
  for (int i = j + 1; i < D_DIM; ++i) {
    float s = 0.0f;
    for (int mm = j; mm < i; ++mm) s += ldsL[i * 65 + mm] * ldsA[j * 65 + mm];
    ldsA[j * 65 + i] = -s / ldsL[i * 65 + i];
  }
  __syncthreads();

  // v_k[j] = sum_m Ainv[j][m] * mu[m]   (Ainv[j][m]=0 for m>j, stored as 0)
  {
    float s = 0.0f;
    for (int mm = 0; mm <= j; ++mm)
      s += ldsA[mm * 65 + j] * mus[k * D_DIM + mm];
    vws[k * D_DIM + j] = s;
  }

  if (j == 0) {
    float logdet = 0.0f;
    for (int i = 0; i < D_DIM; ++i) logdet += logf(ldsL[i * 65 + i]);
    float mx = -INFINITY;
    for (int t = 0; t < K_COMP; ++t) mx = fmaxf(mx, logits[t]);
    float se = 0.0f;
    for (int t = 0; t < K_COMP; ++t) se += expf(logits[t] - mx);
    float lse = mx + logf(se);
    cws[k] = (logits[k] - lse) - logdet - 0.5f * (float)D_DIM * LOG2PI;
  }

  // emit WMMA A-fragments (hi part then lo part)
  for (int idx = j; idx < 8192; idx += 64) {
    int e    = idx & 15;
    int l    = (idx >> 4) & 31;
    int ks   = (idx >> 9) & 1;
    int m    = (idx >> 10) & 3;
    int part = (idx >> 12) & 1;
    int hi   = l >> 4;
    int zrow = 16 * m + (l & 15);
    int kk   = ks * 32 + hi * 16 + e;
    float val = ldsA[kk * 65 + zrow];            // Ainv[zrow][kk]
    unsigned short bh = f32_to_bf16(val);
    unsigned short out = (part == 0) ? bh : f32_to_bf16(val - bf16_to_f32(bh));
    aws[(size_t)k * 8192 + idx] = out;
  }
}

// ---------------------------------------------------------------------------
// Main: 256 threads = 8 waves, each wave owns 16 rows of x (block = 128 rows).
// ---------------------------------------------------------------------------
__global__ __launch_bounds__(256) void gmm_main_kernel(
    const float* __restrict__ x,
    const unsigned short* __restrict__ aws,
    const float* __restrict__ vws,
    const float* __restrict__ cws,
    float* __restrict__ partials) {
  __shared__ float wsum[8];
  const int tid  = threadIdx.x;
  const int lane = tid & 31;
  const int wave = tid >> 5;
  const int l16  = lane & 15;
  const int hi   = lane >> 4;
  const int row  = blockIdx.x * 128 + wave * 16 + l16;

  // Build B fragments (32x16 bf16, contiguous k order: lanes 0-15 hold K=0..15 of col N=lane,
  // lanes 16-31 hold K=16..31) with hi/lo split, for both K-steps of the D=64 dim.
  v16bf Bh[2], Bl[2];
#pragma unroll
  for (int ks = 0; ks < 2; ++ks) {
    const float4* xp4 =
        reinterpret_cast<const float4*>(x + (size_t)row * D_DIM + ks * 32 + hi * 16);
    float xf[16];
#pragma unroll
    for (int q = 0; q < 4; ++q) {
      float4 f = xp4[q];
      xf[q * 4 + 0] = f.x; xf[q * 4 + 1] = f.y;
      xf[q * 4 + 2] = f.z; xf[q * 4 + 3] = f.w;
    }
#pragma unroll
    for (int e = 0; e < 16; ++e) {
      unsigned short h = f32_to_bf16(xf[e]);
      Bh[ks][e] = __builtin_bit_cast(__bf16, h);
      Bl[ks][e] = __builtin_bit_cast(__bf16, f32_to_bf16(xf[e] - bf16_to_f32(h)));
    }
  }

  const v8f vzero = {0.f, 0.f, 0.f, 0.f, 0.f, 0.f, 0.f, 0.f};
  float run_m = -INFINITY, run_s = 0.0f;

  for (int k = 0; k < K_COMP; ++k) {
    const v16bf* afrag = reinterpret_cast<const v16bf*>(aws + (size_t)k * 8192);
    v8f acc[4];
#pragma unroll
    for (int m = 0; m < 4; ++m) acc[m] = vzero;

#pragma unroll
    for (int m = 0; m < 4; ++m) {
#pragma unroll
      for (int ks = 0; ks < 2; ++ks) {
        v16bf Ah = afrag[m * 64 + ks * 32 + lane];         // hi part
        v16bf Al = afrag[256 + m * 64 + ks * 32 + lane];   // lo part
        // 3-term split: Ah*Bh + Ah*Bl + Al*Bh  (~fp32 accuracy)
        acc[m] = __builtin_amdgcn_wmma_f32_16x16x32_bf16(
            false, Ah, false, Bh[ks], (short)0, acc[m], false, false);
        acc[m] = __builtin_amdgcn_wmma_f32_16x16x32_bf16(
            false, Ah, false, Bl[ks], (short)0, acc[m], false, false);
        acc[m] = __builtin_amdgcn_wmma_f32_16x16x32_bf16(
            false, Al, false, Bh[ks], (short)0, acc[m], false, false);
      }
    }

    // maha: lane holds M = r + 8*hi of z-block m for its row N = l16.
    float part = 0.0f;
#pragma unroll
    for (int m = 0; m < 4; ++m) {
      const float* vp = vws + k * D_DIM + 16 * m + 8 * hi;
#pragma unroll
      for (int r = 0; r < 8; ++r) {
        float d = acc[m][r] - vp[r];
        part = fmaf(d, d, part);
      }
    }
    // fold lane <-> lane^16 (SWAPX16): both halves of M-range of the same row.
    float maha = part + swz<0x401F>(part);

    float s = cws[k] - 0.5f * maha;   // comp_lp + mix_lp
    if (s > run_m) {
      run_s = run_s * expf(run_m - s) + 1.0f;
      run_m = s;
    } else {
      run_s += expf(s - run_m);
    }
  }

  float lp = run_m + logf(run_s);   // lanes l and l^16 hold identical lp

  // wave reduction over all 32 lanes (x2 duplication -> scale by 0.5)
  float t = lp;
  t += swz<0x041F>(t);  // xor 1
  t += swz<0x081F>(t);  // xor 2
  t += swz<0x101F>(t);  // xor 4
  t += swz<0x201F>(t);  // xor 8
  t += swz<0x401F>(t);  // xor 16
  if (lane == 0) wsum[wave] = 0.5f * t;
  __syncthreads();
  if (tid == 0) {
    float b = 0.0f;
#pragma unroll
    for (int w = 0; w < 8; ++w) b += wsum[w];
    partials[blockIdx.x] = b;
  }
}

// ---------------------------------------------------------------------------
// Final: deterministic sequential reduction of block partials.
// ---------------------------------------------------------------------------
__global__ void gmm_final_kernel(const float* __restrict__ partials,
                                 float* __restrict__ out, int nblocks,
                                 float invN) {
  if (blockIdx.x == 0 && threadIdx.x == 0) {
    float s = 0.0f;
    for (int i = 0; i < nblocks; ++i) s += partials[i];
    out[0] = -s * invN;
  }
}

extern "C" void kernel_launch(void* const* d_in, const int* in_sizes, int n_in,
                              void* d_out, int out_size, void* d_ws, size_t ws_size,
                              hipStream_t stream) {
  const float* x      = (const float*)d_in[0];
  const float* mus    = (const float*)d_in[1];
  const float* L      = (const float*)d_in[2];
  const float* logits = (const float*)d_in[3];

  const int N = in_sizes[0] / D_DIM;        // 65536
  const int nblocks = N / 128;              // 512

  // workspace carve-up
  unsigned short* aws = (unsigned short*)d_ws;                       // 512 KB
  float* vws      = (float*)((char*)d_ws + 524288);                  // 8 KB
  float* cws      = (float*)((char*)d_ws + 524288 + 8192);           // 128 B
  float* partials = (float*)((char*)d_ws + 524288 + 8192 + 256);     // 2 KB

  gmm_setup_kernel<<<K_COMP, 64, 0, stream>>>(mus, L, logits, aws, vws, cws);
  gmm_main_kernel<<<nblocks, 256, 0, stream>>>(x, aws, vws, cws, partials);
  gmm_final_kernel<<<1, 32, 0, stream>>>(partials, (float*)d_out, nblocks,
                                         1.0f / (float)N);
}
